// GraphEnhancer_9878424780849
// MI455X (gfx1250) — matrix operand order repaired and checked by reading, not compiled, not run on hardware
//
#include <hip/hip_runtime.h>
#include <hip/hip_bf16.h>

// ---------------------------------------------------------------------------
// Problem constants (from reference)
// ---------------------------------------------------------------------------
#define N_NODES 50000
#define N_EDGES 512000
#define N_RELS  8
#define DIM     256
#define HID     256
#define ADAPT   1024
#define OUT_D   4096
#define BATCH   32
#define K_ENT   50
#define N_QE    (BATCH + BATCH * K_ENT)   // 1632 adapter rows
#define KCAT    (N_RELS * DIM + DIM)      // 2304 stacked K for a layer GEMM

// LDS staging geometry (per 128-row x 32-k A tile, 128-col x 32-k B tile)
#define A_ROW_F   36                      // 32 f32 + 4 pad -> 144B rows (16B aligned, bank-spread)
#define B_ROW_H   40                      // 32 bf16 + 8 pad -> 80B rows (16B aligned, bank-spread)
#define A_BUF_F   (128 * A_ROW_F)         // floats per A buffer
#define B_BUF_H   (128 * B_ROW_H)         // ushorts per B buffer

// ---------------------------------------------------------------------------
// WMMA types (gfx1250, wave32)
// ---------------------------------------------------------------------------
typedef __attribute__((ext_vector_type(16))) __bf16 v16bf;
typedef __attribute__((ext_vector_type(8)))  float  v8f;

union FragAB {
    v16bf v;
    unsigned short u[16];
    uint4 q[2];
};
union FragC {
    v8f   v;
    float f[8];
};

__device__ __forceinline__ unsigned short f2bf(float f) {
    // round-to-nearest-even f32 -> bf16
    unsigned int u = __float_as_uint(f);
    u += 0x7fffu + ((u >> 16) & 1u);
    return (unsigned short)(u >> 16);
}

// 16B async copy global -> LDS (ASYNCcnt-tracked; CDNA5 VGLOBAL async op)
__device__ __forceinline__ void async_copy16(unsigned lds_off, const void* gptr) {
    asm volatile("global_load_async_to_lds_b128 %0, %1, off"
                 :: "v"(lds_off), "v"(gptr)
                 : "memory");
}

__device__ __forceinline__ void wait_async0() {
#if __has_builtin(__builtin_amdgcn_s_wait_asynccnt)
    __builtin_amdgcn_s_wait_asynccnt(0);
#else
    asm volatile("s_wait_asynccnt 0x0" ::: "memory");
#endif
}

// ---------------------------------------------------------------------------
// base[n] = emb_table[node_ids[n]]   (float4 coalesced)
// ---------------------------------------------------------------------------
__global__ void gather_base_kernel(const int* __restrict__ ids,
                                   const float4* __restrict__ emb4,
                                   float4* __restrict__ base4) {
    int t = blockIdx.x * blockDim.x + threadIdx.x;       // N_NODES*64 threads
    int n = t >> 6, c = t & 63;
    base4[t] = emb4[(size_t)ids[n] * 64 + c];
}

// ---------------------------------------------------------------------------
// Weight prep: Wt[n*K + k] = bf16( k<K1 ? S1[k][n] : S2[k-K1][n] )
// ---------------------------------------------------------------------------
__global__ void cvt_weights_kernel(const float* __restrict__ s1, int K1,
                                   const float* __restrict__ s2, int K2,
                                   int Nout, unsigned short* __restrict__ wt,
                                   int total) {
    int t = blockIdx.x * blockDim.x + threadIdx.x;
    if (t >= total) return;
    int K = K1 + K2;
    int n = t / K, k = t % K;
    float v = (k < K1) ? s1[(size_t)k * Nout + n]
                       : s2[(size_t)(k - K1) * Nout + n];
    wt[t] = f2bf(v);
}

// ---------------------------------------------------------------------------
// Edge scatter:  AGG[dst, etype*256 + c] += x[src, c]
// ---------------------------------------------------------------------------
__global__ void scatter_edges_kernel(const float4* __restrict__ x4,
                                     const int* __restrict__ src,
                                     const int* __restrict__ dst,
                                     const int* __restrict__ ety,
                                     float* __restrict__ agg) {
    int t = blockIdx.x * blockDim.x + threadIdx.x;       // N_EDGES*64 threads
    int e = t >> 6, c = t & 63;
    float4 v = x4[(size_t)src[e] * 64 + c];
    float* o = agg + (size_t)dst[e] * (size_t)(N_RELS * DIM)
                   + (size_t)ety[e] * DIM + (size_t)c * 4;
    __hip_atomic_fetch_add(o + 0, v.x, __ATOMIC_RELAXED, __HIP_MEMORY_SCOPE_AGENT);
    __hip_atomic_fetch_add(o + 1, v.y, __ATOMIC_RELAXED, __HIP_MEMORY_SCOPE_AGENT);
    __hip_atomic_fetch_add(o + 2, v.z, __ATOMIC_RELAXED, __HIP_MEMORY_SCOPE_AGENT);
    __hip_atomic_fetch_add(o + 3, v.w, __ATOMIC_RELAXED, __HIP_MEMORY_SCOPE_AGENT);
}

// ---------------------------------------------------------------------------
// Qin[row] = concat(base[idx], upd[idx]); rows 0..31 = queries, 32.. = entities
// ---------------------------------------------------------------------------
__global__ void build_qin_kernel(const int* __restrict__ qidx,
                                 const int* __restrict__ eidx,
                                 const float4* __restrict__ base4,
                                 const float4* __restrict__ u4,
                                 float4* __restrict__ qin4) {
    int t = blockIdx.x * blockDim.x + threadIdx.x;       // N_QE*128 threads
    if (t >= N_QE * 128) return;
    int row = t >> 7, c = t & 127;
    int idx = (row < BATCH) ? qidx[row] : eidx[row - BATCH];
    qin4[t] = (c < 64) ? base4[(size_t)idx * 64 + c]
                       : u4[(size_t)idx * 64 + (c - 64)];
}

// ---------------------------------------------------------------------------
// bf16-WMMA GEMM, LDS-staged via async global->LDS copies, double buffered:
//   C[M,N] = act( [A1 | A2][M, K1+K2] @ Bt^T + bias )
// A segments are f32 in memory (converted to bf16 in-register from LDS).
// Bt is bf16 stored [N][K1+K2].  Block = 8 waves = 128 rows x 128 cols;
// wave computes 16 rows x 128 cols (8 wmma tiles, shared A fragment).
// K1,K2 multiples of 32; N multiple of 128. act: 0=none, 1=relu, 2=silu.
// ---------------------------------------------------------------------------
__global__ __launch_bounds__(256, 1)
void gemm_bf16_wmma(const float* __restrict__ A1, int lda1, int K1,
                    const float* __restrict__ A2, int lda2, int K2,
                    const unsigned short* __restrict__ Bt,
                    const float* __restrict__ bias,
                    float* __restrict__ C, int ldc,
                    int M, int act) {
    __shared__ float          shA[2 * A_BUF_F];   // 36,864 B
    __shared__ unsigned short shB[2 * B_BUF_H];   // 20,480 B

    const int tid  = threadIdx.x;
    const int lane = tid & 31;
    const int wave = tid >> 5;
    const int half = lane >> 4;
    const int mn   = lane & 15;
    const int K    = K1 + K2;
    const int rowB = blockIdx.x * 128;
    const int col0 = blockIdx.y * 128;
    const int nsteps = K >> 5;

    // ---- per-thread staging bases (hoisted out of the k loop) ----
    // A tile: 128 rows x 32 f32 = 1024 x 16B chunks; thread covers 4 rows.
    const int arI = tid >> 3;          // 0..31 row sub-index
    const int acI = tid & 7;           // 0..7  16B chunk in row
    const float* aBase1[4];
    const float* aBase2[4];
    unsigned     aLoff[4];
#pragma unroll
    for (int j = 0; j < 4; ++j) {
        const int r = j * 32 + arI;
        int g = rowB + r;
        if (g > M - 1) g = M - 1;      // clamp (clamped rows never stored)
        aBase1[j] = A1 + (size_t)g * lda1 + acI * 4;
        aBase2[j] = A2 ? (A2 + (size_t)g * lda2 + acI * 4) : aBase1[j];
        aLoff[j]  = (unsigned)(size_t)&shA[r * A_ROW_F + acI * 4];
    }
    // B tile: 128 cols x 32 bf16 = 512 x 16B chunks; thread covers 2 cols.
    const int bnI = tid >> 2;          // 0..63 col sub-index
    const int bcI = tid & 3;           // 0..3  16B chunk in row
    const unsigned short* bBase[2];
    unsigned              bLoff[2];
#pragma unroll
    for (int j = 0; j < 2; ++j) {
        const int n = j * 64 + bnI;
        bBase[j] = Bt + (size_t)(col0 + n) * K + bcI * 8;
        bLoff[j] = (unsigned)(size_t)&shB[n * B_ROW_H + bcI * 8];
    }

    auto stage = [&](int s, int buf) {
        const int kb = s << 5;
        const bool seg2 = (kb >= K1);
        const int  kk   = seg2 ? (kb - K1) : kb;
        const unsigned aB = (unsigned)buf * (A_BUF_F * 4);
        const unsigned bB = (unsigned)buf * (B_BUF_H * 2);
#pragma unroll
        for (int j = 0; j < 4; ++j)
            async_copy16(aLoff[j] + aB,
                         (const void*)((seg2 ? aBase2[j] : aBase1[j]) + kk));
#pragma unroll
        for (int j = 0; j < 2; ++j)
            async_copy16(bLoff[j] + bB, (const void*)(bBase[j] + kb));
    };

    v8f acc[8];
#pragma unroll
    for (int t = 0; t < 8; ++t) acc[t] = (v8f){0.f,0.f,0.f,0.f,0.f,0.f,0.f,0.f};

    stage(0, 0);
    for (int s = 0; s < nsteps; ++s) {
        const int cur = s & 1;
        wait_async0();            // my async copies for buffer `cur` landed
        __syncthreads();          // everyone's landed; prev buffer free to refill
        if (s + 1 < nsteps) stage(s + 1, cur ^ 1);

        // A fragment (ISA 16-bit A layout): lane row = wave*16+mn,
        // vgpr v holds k0,k0+1 with k0 = (v>>2)*16 + (v&3)*2 + half*8
        FragAB a;
        const float* ap = &shA[cur * A_BUF_F + (wave * 16 + mn) * A_ROW_F];
#pragma unroll
        for (int v = 0; v < 8; ++v) {
            const int k0 = ((v >> 2) << 4) + ((v & 3) << 1) + (half << 3);
            const float2 f = *(const float2*)(ap + k0);
            a.u[2 * v]     = f2bf(f.x);
            a.u[2 * v + 1] = f2bf(f.y);
        }
        // preload all 8 B fragments, then run the WMMA chain (lets the ds
        // loads pipeline instead of a full dscnt stall before every wmma)
        FragAB b[8];
        const unsigned short* bp0 = &shB[cur * B_BUF_H + half * 8];
#pragma unroll
        for (int t = 0; t < 8; ++t) {
            const unsigned short* bp = bp0 + (t * 16 + mn) * B_ROW_H;
            b[t].q[0] = *(const uint4*)(bp);
            b[t].q[1] = *(const uint4*)(bp + 16);
        }
#pragma unroll
        for (int t = 0; t < 8; ++t)
            acc[t] = __builtin_amdgcn_wmma_f32_16x16x32_bf16(
                false, a.v, false, b[t].v, (short)0, acc[t], false, false);
    }

    // epilogue: C layout vgpr r -> M = r + 8*half, N = mn
    const int row0 = rowB + wave * 16;
#pragma unroll
    for (int t = 0; t < 8; ++t) {
        FragC cc; cc.v = acc[t];
#pragma unroll
        for (int r = 0; r < 8; ++r) {
            const int gm = row0 + r + half * 8;
            const int gn = col0 + t * 16 + mn;
            float v = cc.f[r];
            if (bias) v += bias[gn];
            if (act == 1) v = fmaxf(v, 0.f);
            else if (act == 2) v = v * (1.f / (1.f + __expf(-v)));
            if (gm < M) C[(size_t)gm * ldc + gn] = v;
        }
    }
}

// ---------------------------------------------------------------------------
// Host-side orchestration
// ---------------------------------------------------------------------------
extern "C" void kernel_launch(void* const* d_in, const int* in_sizes, int n_in,
                              void* d_out, int out_size, void* d_ws, size_t ws_size,
                              hipStream_t stream) {
    const int*   qidx = (const int*)d_in[0];
    const int*   eidx = (const int*)d_in[1];
    const int*   ids  = (const int*)d_in[2];
    const int*   src  = (const int*)d_in[3];
    const int*   dst  = (const int*)d_in[4];
    const int*   ety  = (const int*)d_in[5];
    const float* emb  = (const float*)d_in[6];
    const float* W1   = (const float*)d_in[7];
    const float* W1l  = (const float*)d_in[8];
    const float* b1   = (const float*)d_in[9];
    const float* W2   = (const float*)d_in[10];
    const float* W2l  = (const float*)d_in[11];
    const float* b2   = (const float*)d_in[12];
    const float* A1   = (const float*)d_in[13];
    const float* A2   = (const float*)d_in[14];
    float* out = (float*)d_out;
    (void)in_sizes; (void)n_in; (void)out_size; (void)ws_size;

    // workspace layout
    char* ws = (char*)d_ws;
    size_t off = 0;
    auto alloc = [&](size_t bytes) {
        void* p = ws + off;
        off += (bytes + 255) & ~(size_t)255;
        return p;
    };
    float* AGG  = (float*)alloc((size_t)N_NODES * N_RELS * DIM * sizeof(float)); // 409.6MB
    float* BASE = (float*)alloc((size_t)N_NODES * DIM * sizeof(float));
    float* H    = (float*)alloc((size_t)N_NODES * DIM * sizeof(float));
    float* U    = (float*)alloc((size_t)N_NODES * DIM * sizeof(float));
    unsigned short* Wt1 = (unsigned short*)alloc((size_t)HID   * KCAT  * 2);
    unsigned short* Wt2 = (unsigned short*)alloc((size_t)DIM   * KCAT  * 2);
    unsigned short* A1t = (unsigned short*)alloc((size_t)ADAPT * 2 * DIM * 2);
    unsigned short* A2t = (unsigned short*)alloc((size_t)OUT_D * ADAPT * 2);
    float* QIN = (float*)alloc((size_t)N_QE * 2 * DIM * sizeof(float));
    float* Y1  = (float*)alloc((size_t)N_QE * ADAPT * sizeof(float));

    const int TB = 256;

    // 0) weight prep (bf16, transposed [N][K])
    {
        int tot = HID * KCAT;
        cvt_weights_kernel<<<(tot + TB - 1) / TB, TB, 0, stream>>>(
            W1, N_RELS * DIM, W1l, DIM, HID, Wt1, tot);
        cvt_weights_kernel<<<(tot + TB - 1) / TB, TB, 0, stream>>>(
            W2, N_RELS * HID, W2l, HID, DIM, Wt2, tot);
        int tot1 = ADAPT * (2 * DIM);
        cvt_weights_kernel<<<(tot1 + TB - 1) / TB, TB, 0, stream>>>(
            A1, 2 * DIM, nullptr, 0, ADAPT, A1t, tot1);
        int tot2 = OUT_D * ADAPT;
        cvt_weights_kernel<<<(tot2 + TB - 1) / TB, TB, 0, stream>>>(
            A2, ADAPT, nullptr, 0, OUT_D, A2t, tot2);
    }

    // 1) base = emb_table[node_ids]
    gather_base_kernel<<<(N_NODES * 64) / TB, TB, 0, stream>>>(
        ids, (const float4*)emb, (float4*)BASE);

    const dim3 gemm_blk(256);
    const dim3 grid_l((N_NODES + 127) / 128, DIM / 128);    // 391 x 2
    const dim3 grid_a1((N_QE + 127) / 128, ADAPT / 128);    // 13 x 8
    const dim3 grid_a2((N_QE + 127) / 128, OUT_D / 128);    // 13 x 32

    // 2) layer 1: AGG = scatter(base);  H = relu([AGG|base] @ Wt1 + b1)
    hipMemsetAsync(AGG, 0, (size_t)N_NODES * N_RELS * DIM * sizeof(float), stream);
    scatter_edges_kernel<<<(N_EDGES * 64) / TB, TB, 0, stream>>>(
        (const float4*)BASE, src, dst, ety, AGG);
    gemm_bf16_wmma<<<grid_l, gemm_blk, 0, stream>>>(
        AGG, N_RELS * DIM, N_RELS * DIM, BASE, DIM, DIM,
        Wt1, b1, H, HID, N_NODES, /*act=relu*/1);

    // 3) layer 2: AGG = scatter(H);  U = [AGG|H] @ Wt2 + b2
    hipMemsetAsync(AGG, 0, (size_t)N_NODES * N_RELS * DIM * sizeof(float), stream);
    scatter_edges_kernel<<<(N_EDGES * 64) / TB, TB, 0, stream>>>(
        (const float4*)H, src, dst, ety, AGG);
    gemm_bf16_wmma<<<grid_l, gemm_blk, 0, stream>>>(
        AGG, N_RELS * HID, N_RELS * HID, H, HID, HID,
        Wt2, b2, U, DIM, N_NODES, /*act=none*/0);

    // 4) adapter input rows: [query(32) ; entities(1600)] x [base | upd]
    {
        int tot = N_QE * 128;
        build_qin_kernel<<<(tot + TB - 1) / TB, TB, 0, stream>>>(
            qidx, eidx, (const float4*)BASE, (const float4*)U, (float4*)QIN);
    }

    // 5) Y1 = silu(QIN @ A1t)
    gemm_bf16_wmma<<<grid_a1, gemm_blk, 0, stream>>>(
        QIN, 2 * DIM, 2 * DIM, nullptr, 0, 0,
        A1t, nullptr, Y1, ADAPT, N_QE, /*act=silu*/2);

    // 6) out = Y1 @ A2t   (rows 0..31 -> query_embeds, 32..1631 -> entity_embeds)
    gemm_bf16_wmma<<<grid_a2, gemm_blk, 0, stream>>>(
        Y1, ADAPT, ADAPT, nullptr, 0, 0,
        A2t, nullptr, out, OUT_D, N_QE, /*act=none*/0);
}
